// DeepFM_56023553409246
// MI455X (gfx1250) — matrix operand order repaired and checked by reading, not compiled, not run on hardware
//
#include <hip/hip_runtime.h>
#include <math.h>

typedef __attribute__((ext_vector_type(16))) _Float16 v16h;
typedef __attribute__((ext_vector_type(8)))  _Float16 v8h;
typedef __attribute__((ext_vector_type(8)))  float    v8f;

#define B_ROWS 4096
#define NFIELD 26
#define EMB_D  16
#define VOCAB  200000
#define MLP_IN 416      // 26*16 = 13 k-tiles of 32
#define H1     256      // 8 k-tiles of 32
#define H2     128
#define EPS    1e-5f

// ---------------------------------------------------------------------------
// 1) Pack W [N,K] f32 row-major -> f16 WMMA B-fragments.
//    Fragment (kt,nt): 32 lanes x 16 halfs, lane<16 holds col n=nt*16+lane,
//    K = kt*32 + 0..15; lane>=16 holds same col, K = kt*32 + 16..31.
// ---------------------------------------------------------------------------
__global__ void pack_w_kernel(const float* __restrict__ W, _Float16* __restrict__ wp,
                              int N, int K) {
    int t = blockIdx.x * blockDim.x + threadIdx.x;
    if (t >= N * K) return;
    int j    = t & 15;          // half index within lane
    int lane = (t >> 4) & 31;
    int frag = t >> 9;          // = kt*NT + nt
    int NT   = N >> 4;
    int kt   = frag / NT;
    int nt   = frag - kt * NT;
    int n    = nt * 16 + (lane & 15);
    int k    = kt * 32 + ((lane >> 4) << 4) + j;
    wp[t] = (_Float16)W[(size_t)n * K + k];   // B[k][n] = W[n][k]
}

// ---------------------------------------------------------------------------
// 2) Embedding gather + FM part. Thread = (row b, dim d). Wave32: lanes 0-15
//    are row b, lanes 16-31 are row b+1; shfl_xor masks < 16 stay per-row.
// ---------------------------------------------------------------------------
__global__ void gather_fm_kernel(const int* __restrict__ x,
                                 const float* __restrict__ emb,
                                 const float* __restrict__ lin,
                                 const float* __restrict__ fmb,
                                 _Float16* __restrict__ h0,   // [B, 416] f16
                                 float* __restrict__ fm) {    // [B]
    int tid = blockIdx.x * blockDim.x + threadIdx.x;
    int b = tid >> 4;
    int d = tid & 15;
    const int* xb = x + b * NFIELD;
    float s = 0.f, ss = 0.f;
    #pragma unroll
    for (int f = 0; f < NFIELD; ++f) {
        int idx = xb[f];
        float e = emb[((size_t)f * VOCAB + idx) * EMB_D + d];
        s += e; ss += e * e;
        h0[(size_t)b * MLP_IN + f * EMB_D + d] = (_Float16)e;
    }
    // linear terms: lane d covers fields d and d+16 (26 fields total)
    float la = lin[(size_t)d * VOCAB + xb[d]];
    if (d < NFIELD - 16) la += lin[(size_t)(d + 16) * VOCAB + xb[d + 16]];
    float inner = s * s - ss;
    #pragma unroll
    for (int off = 8; off >= 1; off >>= 1) {
        inner += __shfl_xor(inner, off, 32);
        la    += __shfl_xor(la,    off, 32);
    }
    if (d == 0) fm[b] = la + fmb[0] + 0.5f * inner;
}

// ---------------------------------------------------------------------------
// 3) WMMA GEMM: C[M,N] = A[M,K](f16,row-major) * B + bias, B pre-packed.
//    8 waves/block; wave -> one 16-row M-tile; blockIdx.y -> 64-column strip
//    (4 accumulators). v_wmma_f32_16x16x32_f16, f32 accumulate.
// ---------------------------------------------------------------------------
__global__ void __launch_bounds__(256)
gemm_wmma_kernel(const _Float16* __restrict__ A,
                 const _Float16* __restrict__ Wp,
                 const float* __restrict__ bias,
                 float* __restrict__ C,
                 int M, int N, int K) {
    const int lane   = threadIdx.x & 31;
    const int wave   = threadIdx.x >> 5;
    const int mt     = blockIdx.x * 8 + wave;
    const int ntbase = blockIdx.y * 4;
    const int NT = N >> 4;
    const int KT = K >> 5;
    const int m    = mt * 16 + (lane & 15);
    const int khi  = (lane >> 4) << 4;     // lane>=16 starts at K+16 (halfs)

    v8f acc[4];
    #pragma unroll
    for (int i = 0; i < 4; ++i) {
        float bv = bias[(ntbase + i) * 16 + (lane & 15)];
        #pragma unroll
        for (int r = 0; r < 8; ++r) acc[i][r] = bv;
    }

    const _Float16* arow = A + (size_t)m * K;
    for (int kt = 0; kt < KT; ++kt) {
        union { v16h v; v8h h[2]; } a;
        const _Float16* pa = arow + kt * 32 + khi;
        a.h[0] = *(const v8h*)pa;            // K base+0..7   (contiguous, matches packed B)
        a.h[1] = *(const v8h*)(pa + 8);      // K base+8..15
        if (kt + 1 < KT) __builtin_prefetch(arow + (kt + 1) * 32 + khi, 0, 1);
        #pragma unroll
        for (int i = 0; i < 4; ++i) {
            v16h bf = *(const v16h*)(Wp + ((size_t)(kt * NT + ntbase + i) * 32 + lane) * 16);
            acc[i] = __builtin_amdgcn_wmma_f32_16x16x32_f16(
                         false, a.v, false, bf, (short)0, acc[i], false, false);
        }
    }

    // C/D layout: VGPR r -> row mt*16 + r (+8 for lanes 16-31), col = lane&15
    const int rowbase = mt * 16 + ((lane >> 4) << 3);
    #pragma unroll
    for (int i = 0; i < 4; ++i) {
        int col = (ntbase + i) * 16 + (lane & 15);
        #pragma unroll
        for (int r = 0; r < 8; ++r)
            C[(size_t)(rowbase + r) * N + col] = acc[i][r];
    }
}

// ---------------------------------------------------------------------------
// 4) Per-column batch statistics (training-mode biased variance)
// ---------------------------------------------------------------------------
__global__ void colstats_kernel(const float* __restrict__ Z,
                                float* __restrict__ mean,
                                float* __restrict__ invstd, int N) {
    __shared__ float ssum[256], ssq[256];
    int col = blockIdx.x;
    float s = 0.f, q = 0.f;
    for (int r = threadIdx.x; r < B_ROWS; r += 256) {
        float z = Z[(size_t)r * N + col];
        s += z; q += z * z;
    }
    ssum[threadIdx.x] = s; ssq[threadIdx.x] = q;
    __syncthreads();
    for (int off = 128; off >= 1; off >>= 1) {
        if (threadIdx.x < off) {
            ssum[threadIdx.x] += ssum[threadIdx.x + off];
            ssq[threadIdx.x]  += ssq[threadIdx.x + off];
        }
        __syncthreads();
    }
    if (threadIdx.x == 0) {
        float m = ssum[0] * (1.f / B_ROWS);
        float v = ssq[0] * (1.f / B_ROWS) - m * m;
        mean[col]   = m;
        invstd[col] = rsqrtf(v + EPS);
    }
}

// ---------------------------------------------------------------------------
// 5) BN + ReLU -> f16 (next GEMM's A matrix). N must be a power of two.
// ---------------------------------------------------------------------------
__global__ void bn_relu_f16_kernel(const float* __restrict__ Z,
                                   const float* __restrict__ mean,
                                   const float* __restrict__ invstd,
                                   const float* __restrict__ g,
                                   const float* __restrict__ be,
                                   _Float16* __restrict__ H, int N) {
    int t = blockIdx.x * blockDim.x + threadIdx.x;
    int col = t & (N - 1);
    float h = (Z[t] - mean[col]) * invstd[col] * g[col] + be[col];
    H[t] = (_Float16)fmaxf(h, 0.f);
}

// ---------------------------------------------------------------------------
// 6) BN2 + ReLU fused into the final 128-wide dot + sigmoid. 32 lanes / row.
// ---------------------------------------------------------------------------
__global__ void final_kernel(const float* __restrict__ Z2,
                             const float* __restrict__ mean2,
                             const float* __restrict__ invstd2,
                             const float* __restrict__ g2,
                             const float* __restrict__ be2,
                             const float* __restrict__ W3,
                             const float* __restrict__ b3,
                             const float* __restrict__ fm,
                             float* __restrict__ out) {
    int lane = threadIdx.x & 31;
    int b = blockIdx.x * 8 + (threadIdx.x >> 5);
    float p = 0.f;
    #pragma unroll
    for (int j = 0; j < 4; ++j) {
        int col = lane + 32 * j;
        float z = Z2[(size_t)b * H2 + col];
        float h = fmaxf((z - mean2[col]) * invstd2[col] * g2[col] + be2[col], 0.f);
        p += h * W3[col];
    }
    #pragma unroll
    for (int off = 16; off >= 1; off >>= 1) p += __shfl_xor(p, off, 32);
    if (lane == 0) {
        float v = p + b3[0] + fm[b];
        out[b] = 1.f / (1.f + expf(-v));
    }
}

// ---------------------------------------------------------------------------
extern "C" void kernel_launch(void* const* d_in, const int* in_sizes, int n_in,
                              void* d_out, int out_size, void* d_ws, size_t ws_size,
                              hipStream_t stream) {
    (void)in_sizes; (void)n_in; (void)out_size; (void)ws_size;
    const int*   x   = (const int*)  d_in[0];
    const float* emb = (const float*)d_in[1];
    const float* lin = (const float*)d_in[2];
    const float* fmb = (const float*)d_in[3];
    const float* W1  = (const float*)d_in[4];
    const float* b1  = (const float*)d_in[5];
    const float* g1  = (const float*)d_in[6];
    const float* be1 = (const float*)d_in[7];
    const float* W2  = (const float*)d_in[8];
    const float* b2  = (const float*)d_in[9];
    const float* g2  = (const float*)d_in[10];
    const float* be2 = (const float*)d_in[11];
    const float* W3  = (const float*)d_in[12];
    const float* b3  = (const float*)d_in[13];
    float* out = (float*)d_out;

    char* ws = (char*)d_ws;
    size_t off = 0;
    auto alloc = [&](size_t bytes) {
        char* p = ws + off;
        off = (off + bytes + 255) & ~(size_t)255;
        return p;
    };
    _Float16* h0    = (_Float16*)alloc((size_t)B_ROWS * MLP_IN * 2);
    _Float16* wp1   = (_Float16*)alloc((size_t)H1 * MLP_IN * 2);
    _Float16* wp2   = (_Float16*)alloc((size_t)H2 * H1 * 2);
    float*    z1    = (float*)   alloc((size_t)B_ROWS * H1 * 4);
    _Float16* h1    = (_Float16*)alloc((size_t)B_ROWS * H1 * 2);
    float*    z2    = (float*)   alloc((size_t)B_ROWS * H2 * 4);
    float*    fm    = (float*)   alloc((size_t)B_ROWS * 4);
    float*    mean1 = (float*)   alloc(H1 * 4);
    float*    istd1 = (float*)   alloc(H1 * 4);
    float*    mean2 = (float*)   alloc(H2 * 4);
    float*    istd2 = (float*)   alloc(H2 * 4);

    // 1) pack weights into WMMA B-fragment layout (f16)
    pack_w_kernel<<<(H1 * MLP_IN + 255) / 256, 256, 0, stream>>>(W1, wp1, H1, MLP_IN);
    pack_w_kernel<<<(H2 * H1 + 255) / 256, 256, 0, stream>>>(W2, wp2, H2, H1);

    // 2) gather + FM part + f16 activation matrix
    gather_fm_kernel<<<B_ROWS * EMB_D / 256, 256, 0, stream>>>(x, emb, lin, fmb, h0, fm);

    // 3) layer 1 GEMM: [4096,416] x [416,256]
    gemm_wmma_kernel<<<dim3(B_ROWS / 128, H1 / 64), 256, 0, stream>>>(
        h0, wp1, b1, z1, B_ROWS, H1, MLP_IN);
    colstats_kernel<<<H1, 256, 0, stream>>>(z1, mean1, istd1, H1);
    bn_relu_f16_kernel<<<B_ROWS * H1 / 256, 256, 0, stream>>>(z1, mean1, istd1, g1, be1, h1, H1);

    // 4) layer 2 GEMM: [4096,256] x [256,128]
    gemm_wmma_kernel<<<dim3(B_ROWS / 128, H2 / 64), 256, 0, stream>>>(
        h1, wp2, b2, z2, B_ROWS, H2, H1);
    colstats_kernel<<<H2, 256, 0, stream>>>(z2, mean2, istd2, H2);

    // 5) fused BN2 + ReLU + W3 dot + fm + sigmoid
    final_kernel<<<B_ROWS / 8, 256, 0, stream>>>(z2, mean2, istd2, g2, be2, W3, b3, fm, out);
}
